// MultiHeadAttention_46729244180577
// MI455X (gfx1250) — matrix operand order repaired and checked by reading, not compiled, not run on hardware
//
#include <hip/hip_runtime.h>

// ---------------------------------------------------------------------------
// MultiHeadAttention forward for MI455X (gfx1250, wave32)
// B=8, T=1024, DIM=768, H=12, Dh=64
// bf16 v_wmma_f32_16x16x32_bf16 everywhere; tile movement uses the CDNA5
// async Global->LDS DMA path (global_load_async_to_lds_b128 / ASYNCcnt) with
// double-buffered LDS so DMA overlaps WMMA compute.
// ---------------------------------------------------------------------------

typedef __attribute__((ext_vector_type(16))) __bf16 v16bf;
typedef __attribute__((ext_vector_type(8)))  __bf16 v8bf;
typedef __attribute__((ext_vector_type(8)))  float  v8f;

union FragA { v16bf v; v8bf h[2]; };

#define DIMD   768
#define NQKV   2304
#define BATCH  8
#define SEQ    1024
#define NHEAD  12
#define DHEAD  64
#define BHTOT  (BATCH*NHEAD)      // 96
#define MTOT   (BATCH*SEQ)        // 8192

// Async DMA: copy 16 bytes global -> LDS (per enabled lane). LDS offset is the
// low 32 bits of the generic pointer (aperture maps addr[31:0] to LDS).
__device__ __forceinline__ void async_copy16(const void* g, void* l) {
  asm volatile("global_load_async_to_lds_b128 %0, %1, off"
               :: "v"((unsigned)(size_t)l), "v"(g)
               : "memory");
}
#define WAIT_ASYNC(n) asm volatile("s_wait_asynccnt " #n ::: "memory")

// ---------------------------------------------------------------------------
// f32 -> bf16 elementwise convert, 8 elements/thread
__global__ void k_cvt_bf16(const float* __restrict__ src,
                           __bf16* __restrict__ dst, int n8) {
  int i = blockIdx.x * blockDim.x + threadIdx.x;
  if (i >= n8) return;
  const float4* s = (const float4*)src + (size_t)i * 2;
  float4 a = s[0], b = s[1];
  v8bf o;
  o[0] = (__bf16)a.x; o[1] = (__bf16)a.y; o[2] = (__bf16)a.z; o[3] = (__bf16)a.w;
  o[4] = (__bf16)b.x; o[5] = (__bf16)b.y; o[6] = (__bf16)b.z; o[7] = (__bf16)b.w;
  *((v8bf*)dst + i) = o;
}

// ---------------------------------------------------------------------------
// WT[n][k] = (bf16)W[k][n], W is K x N row-major. 32x32 tiles, 256 threads.
__global__ void k_transpose_bf16(const float* __restrict__ W,
                                 __bf16* __restrict__ WT, int K, int N) {
  __shared__ float tile[32][33];
  int n0 = blockIdx.x * 32, k0 = blockIdx.y * 32;
  int tx = threadIdx.x & 31, ty = threadIdx.x >> 5;  // 32 x 8
#pragma unroll
  for (int it = 0; it < 4; ++it)
    tile[ty + 8 * it][tx] = W[(size_t)(k0 + ty + 8 * it) * N + n0 + tx];
  __syncthreads();
#pragma unroll
  for (int it = 0; it < 4; ++it)
    WT[(size_t)(n0 + ty + 8 * it) * K + k0 + tx] = (__bf16)tile[tx][ty + 8 * it];
}

// ---------------------------------------------------------------------------
// QKV GEMM: C(8192 x 2304) = xb(8192 x 768) * Wqkv + bqkv, scattered into
// Q (B,H,T,Dh), K (B,H,T,Dh), Vt (B,H,Dh,T) as bf16.
// Column n of qkv maps to (d = n/36, kk = (n%36)/12, h = n%12).
// 128x128 tile / WG; 8 waves as 4(M) x 2(N); wave tile 32x64 = 2x4 frags.
// Double-buffered LDS fed by async Global->LDS DMA.
__global__ __launch_bounds__(256) void k_qkv_gemm(
    const __bf16* __restrict__ A,    // xb   [8192][768]
    const __bf16* __restrict__ BT,   // WqkvT[2304][768]
    const float* __restrict__ bias,  // [2304]
    __bf16* __restrict__ Qb, __bf16* __restrict__ Kb, __bf16* __restrict__ Vt) {
  __shared__ alignas(32) __bf16 sA[2][128 * 32];
  __shared__ alignas(32) __bf16 sB[2][128 * 32];
  const int m0 = blockIdx.y * 128;
  const int n0 = blockIdx.x * 128;
  const int tid = threadIdx.x;
  const int lane = tid & 31, w = tid >> 5;
  const int l16 = lane & 15, half = lane >> 4;
  const int wm = (w & 3) * 32;   // wave M offset in tile
  const int wn = (w >> 2) * 64;  // wave N offset in tile

  const int lr = tid >> 2;            // 0..63 (row for this thread's pieces)
  const int lc = (tid & 3) * 8;       // col-of-8

  v8f acc[2][4];
#pragma unroll
  for (int mi = 0; mi < 2; ++mi)
#pragma unroll
    for (int ni = 0; ni < 4; ++ni)
#pragma unroll
      for (int e = 0; e < 8; ++e) acc[mi][ni][e] = 0.0f;

  auto issue_tiles = [&](int k0, int buf) {
    async_copy16(&A[(size_t)(m0 + lr) * DIMD + k0 + lc],       &sA[buf][lr * 32 + lc]);
    async_copy16(&A[(size_t)(m0 + lr + 64) * DIMD + k0 + lc],  &sA[buf][(lr + 64) * 32 + lc]);
    async_copy16(&BT[(size_t)(n0 + lr) * DIMD + k0 + lc],      &sB[buf][lr * 32 + lc]);
    async_copy16(&BT[(size_t)(n0 + lr + 64) * DIMD + k0 + lc], &sB[buf][(lr + 64) * 32 + lc]);
  };

  issue_tiles(0, 0);
#pragma unroll 2
  for (int kk = 0; kk < 24; ++kk) {
    const int cur = kk & 1;
    if (kk < 23) {
      issue_tiles((kk + 1) * 32, cur ^ 1);
      WAIT_ASYNC(0x4);               // drain current stage (in-order completion)
    } else {
      WAIT_ASYNC(0x0);
    }
    __syncthreads();

    FragA a[2];
    v16bf b[4];
#pragma unroll
    for (int mi = 0; mi < 2; ++mi) {
      const __bf16* p = &sA[cur][(wm + mi * 16 + l16) * 32 + half * 16];
      a[mi].h[0] = *(const v8bf*)p;
      a[mi].h[1] = *(const v8bf*)(p + 8);
    }
#pragma unroll
    for (int ni = 0; ni < 4; ++ni)
      b[ni] = *(const v16bf*)&sB[cur][(wn + ni * 16 + l16) * 32 + half * 16];
#pragma unroll
    for (int mi = 0; mi < 2; ++mi)
#pragma unroll
      for (int ni = 0; ni < 4; ++ni)
        acc[mi][ni] = __builtin_amdgcn_wmma_f32_16x16x32_bf16(
            false, a[mi].v, false, b[ni], (short)0, acc[mi][ni], false, false);
    __syncthreads();
  }

#pragma unroll
  for (int ni = 0; ni < 4; ++ni) {
    int n = n0 + wn + ni * 16 + l16;
    float bv = bias[n];
    int d = n / 36, rem = n % 36;
    int kk = rem / 12, h = rem % 12;
#pragma unroll
    for (int mi = 0; mi < 2; ++mi) {
#pragma unroll
      for (int r = 0; r < 8; ++r) {
        int m = m0 + wm + mi * 16 + r + half * 8;
        int bidx = m >> 10, t = m & 1023;
        int bh = bidx * NHEAD + h;
        __bf16 val = (__bf16)(acc[mi][ni][r] + bv);
        if (kk == 0)
          Qb[((size_t)bh * SEQ + t) * DHEAD + d] = val;
        else if (kk == 1)
          Kb[((size_t)bh * SEQ + t) * DHEAD + d] = val;
        else
          Vt[((size_t)bh * DHEAD + d) * SEQ + t] = val;
      }
    }
  }
}

// ---------------------------------------------------------------------------
// Flash attention: one WG handles 128 Q rows of one (b,h); 8 waves x 16 rows.
// Each wave owns the full 32-token chunk width, so softmax stats reduce with
// 16-lane shuffles. K/V chunks double-buffered via async Global->LDS DMA.
// NOTE: reference has no 1/sqrt(d) scaling.
__global__ __launch_bounds__(256) void k_attn(
    const __bf16* __restrict__ Qb, const __bf16* __restrict__ Kb,
    const __bf16* __restrict__ Vt, __bf16* __restrict__ Ob /* [8192][768] */) {
  __shared__ alignas(32) __bf16 sK[2][32 * 64];     // [token][d]
  __shared__ alignas(32) __bf16 sV[2][64 * 32];     // [d][token]
  __shared__ alignas(32) __bf16 sP[8 * 16 * 32];    // per-wave P tiles

  const int qt = blockIdx.x & 7;
  const int bh = blockIdx.x >> 3;
  const int bidx = bh / NHEAD, h = bh % NHEAD;
  const int tid = threadIdx.x, lane = tid & 31, w = tid >> 5;
  const int l16 = lane & 15, half = lane >> 4;

  const __bf16* Qg = Qb + (size_t)bh * SEQ * DHEAD;
  const __bf16* Kg = Kb + (size_t)bh * SEQ * DHEAD;
  const __bf16* Vg = Vt + (size_t)bh * DHEAD * SEQ;
  const int qrow0 = qt * 128 + w * 16;

  // Preload this wave's Q A-fragments (reused for all 32 K/V chunks).
  FragA qa[2];
#pragma unroll
  for (int ks = 0; ks < 2; ++ks) {
    const __bf16* p = &Qg[(size_t)(qrow0 + l16) * DHEAD + ks * 32 + half * 16];
    qa[ks].h[0] = *(const v8bf*)p;
    qa[ks].h[1] = *(const v8bf*)(p + 8);
  }

  v8f o[4];
#pragma unroll
  for (int nf = 0; nf < 4; ++nf)
#pragma unroll
    for (int e = 0; e < 8; ++e) o[nf][e] = 0.0f;
  float rmax[8], rsum[8];
#pragma unroll
  for (int r = 0; r < 8; ++r) { rmax[r] = -__builtin_inff(); rsum[r] = 0.0f; }

  const int kr = tid >> 3, kc = (tid & 7) * 8;  // K chunk piece: 32 x 64
  const int vr = tid >> 2, vc = (tid & 3) * 8;  // V chunk piece: 64 x 32
  auto issue_chunk = [&](int j, int buf) {
    const int tb = j * 32;
    async_copy16(&Kg[(size_t)(tb + kr) * DHEAD + kc], &sK[buf][kr * 64 + kc]);
    async_copy16(&Vg[(size_t)vr * SEQ + tb + vc],     &sV[buf][vr * 32 + vc]);
  };

  issue_chunk(0, 0);
#pragma unroll 2
  for (int j = 0; j < 32; ++j) {
    const int cur = j & 1;
    if (j < 31) {
      issue_chunk(j + 1, cur ^ 1);
      WAIT_ASYNC(0x2);
    } else {
      WAIT_ASYNC(0x0);
    }
    __syncthreads();

    // S = Q * K^T  (2 N-frags x 2 K-steps = 4 WMMA)
    v8f s[2];
#pragma unroll
    for (int nf = 0; nf < 2; ++nf) {
      v8f sv = {0.f, 0.f, 0.f, 0.f, 0.f, 0.f, 0.f, 0.f};
#pragma unroll
      for (int ks = 0; ks < 2; ++ks) {
        v16bf bk = *(const v16bf*)&sK[cur][(nf * 16 + l16) * 64 + ks * 32 + half * 16];
        sv = __builtin_amdgcn_wmma_f32_16x16x32_bf16(
            false, qa[ks].v, false, bk, (short)0, sv, false, false);
      }
      s[nf] = sv;
    }

    // Online softmax update; write P (bf16) into wave-private LDS tile.
#pragma unroll
    for (int r = 0; r < 8; ++r) {
      float curm = fmaxf(s[0][r], s[1][r]);
      curm = fmaxf(curm, __shfl_xor(curm, 1));
      curm = fmaxf(curm, __shfl_xor(curm, 2));
      curm = fmaxf(curm, __shfl_xor(curm, 4));
      curm = fmaxf(curm, __shfl_xor(curm, 8));
      float nm = fmaxf(rmax[r], curm);
      float alpha = __expf(rmax[r] - nm);
      rmax[r] = nm;
      float p0 = __expf(s[0][r] - nm);
      float p1 = __expf(s[1][r] - nm);
      float ps = p0 + p1;
      ps += __shfl_xor(ps, 1);
      ps += __shfl_xor(ps, 2);
      ps += __shfl_xor(ps, 4);
      ps += __shfl_xor(ps, 8);
      rsum[r] = rsum[r] * alpha + ps;
#pragma unroll
      for (int nf = 0; nf < 4; ++nf) o[nf][r] *= alpha;
      __bf16* pp = &sP[(w * 16 + r + half * 8) * 32 + l16];
      pp[0]  = (__bf16)p0;
      pp[16] = (__bf16)p1;
    }

    // Reload P as A-fragment (same-wave LDS ops are in order).
    FragA pa;
    const __bf16* pl = &sP[(w * 16 + l16) * 32 + half * 16];
    pa.h[0] = *(const v8bf*)pl;
    pa.h[1] = *(const v8bf*)(pl + 8);

    // O += P * V  (4 N-frags over Dh, single K-step of 32 tokens)
#pragma unroll
    for (int nf = 0; nf < 4; ++nf) {
      v16bf bv = *(const v16bf*)&sV[cur][(nf * 16 + l16) * 32 + half * 16];
      o[nf] = __builtin_amdgcn_wmma_f32_16x16x32_bf16(
          false, pa.v, false, bv, (short)0, o[nf], false, false);
    }
    __syncthreads();
  }

  // Normalize and store to attn buffer at [b, t, h*64 + d] (bf16).
#pragma unroll
  for (int r = 0; r < 8; ++r) {
    float inv = 1.0f / rsum[r];
    int t = qrow0 + r + half * 8;
    size_t rowoff = ((size_t)bidx * SEQ + t) * DIMD + h * DHEAD;
#pragma unroll
    for (int nf = 0; nf < 4; ++nf)
      Ob[rowoff + nf * 16 + l16] = (__bf16)(o[nf][r] * inv);
  }
}

// ---------------------------------------------------------------------------
// Output projection: out(8192 x 768 f32) = attn(bf16) * Wout + bout.
__global__ __launch_bounds__(256) void k_out_gemm(
    const __bf16* __restrict__ A,    // attn [8192][768]
    const __bf16* __restrict__ BT,   // WoutT [768][768]
    const float* __restrict__ bias,  // [768]
    float* __restrict__ C) {         // [8192][768]
  __shared__ alignas(32) __bf16 sA[2][128 * 32];
  __shared__ alignas(32) __bf16 sB[2][128 * 32];
  const int m0 = blockIdx.y * 128;
  const int n0 = blockIdx.x * 128;
  const int tid = threadIdx.x;
  const int lane = tid & 31, w = tid >> 5;
  const int l16 = lane & 15, half = lane >> 4;
  const int wm = (w & 3) * 32;
  const int wn = (w >> 2) * 64;
  const int lr = tid >> 2;
  const int lc = (tid & 3) * 8;

  v8f acc[2][4];
#pragma unroll
  for (int mi = 0; mi < 2; ++mi)
#pragma unroll
    for (int ni = 0; ni < 4; ++ni)
#pragma unroll
      for (int e = 0; e < 8; ++e) acc[mi][ni][e] = 0.0f;

  auto issue_tiles = [&](int k0, int buf) {
    async_copy16(&A[(size_t)(m0 + lr) * DIMD + k0 + lc],       &sA[buf][lr * 32 + lc]);
    async_copy16(&A[(size_t)(m0 + lr + 64) * DIMD + k0 + lc],  &sA[buf][(lr + 64) * 32 + lc]);
    async_copy16(&BT[(size_t)(n0 + lr) * DIMD + k0 + lc],      &sB[buf][lr * 32 + lc]);
    async_copy16(&BT[(size_t)(n0 + lr + 64) * DIMD + k0 + lc], &sB[buf][(lr + 64) * 32 + lc]);
  };

  issue_tiles(0, 0);
#pragma unroll 2
  for (int kk = 0; kk < 24; ++kk) {
    const int cur = kk & 1;
    if (kk < 23) {
      issue_tiles((kk + 1) * 32, cur ^ 1);
      WAIT_ASYNC(0x4);
    } else {
      WAIT_ASYNC(0x0);
    }
    __syncthreads();

    FragA a[2];
    v16bf b[4];
#pragma unroll
    for (int mi = 0; mi < 2; ++mi) {
      const __bf16* p = &sA[cur][(wm + mi * 16 + l16) * 32 + half * 16];
      a[mi].h[0] = *(const v8bf*)p;
      a[mi].h[1] = *(const v8bf*)(p + 8);
    }
#pragma unroll
    for (int ni = 0; ni < 4; ++ni)
      b[ni] = *(const v16bf*)&sB[cur][(wn + ni * 16 + l16) * 32 + half * 16];
#pragma unroll
    for (int mi = 0; mi < 2; ++mi)
#pragma unroll
      for (int ni = 0; ni < 4; ++ni)
        acc[mi][ni] = __builtin_amdgcn_wmma_f32_16x16x32_bf16(
            false, a[mi].v, false, b[ni], (short)0, acc[mi][ni], false, false);
    __syncthreads();
  }

#pragma unroll
  for (int ni = 0; ni < 4; ++ni) {
    int n = n0 + wn + ni * 16 + l16;
    float bv = bias[n];
#pragma unroll
    for (int mi = 0; mi < 2; ++mi)
#pragma unroll
      for (int r = 0; r < 8; ++r) {
        int m = m0 + wm + mi * 16 + r + half * 8;
        C[(size_t)m * DIMD + n] = acc[mi][ni][r] + bv;
      }
  }
}

// ---------------------------------------------------------------------------
extern "C" void kernel_launch(void* const* d_in, const int* in_sizes, int n_in,
                              void* d_out, int out_size, void* d_ws,
                              size_t ws_size, hipStream_t stream) {
  const float* x    = (const float*)d_in[0];
  const float* Wqkv = (const float*)d_in[1];
  const float* bqkv = (const float*)d_in[2];
  const float* Wout = (const float*)d_in[3];
  const float* bout = (const float*)d_in[4];
  float* out = (float*)d_out;

  char* ws = (char*)d_ws;
  size_t off = 0;
  auto wsalloc = [&](size_t bytes) -> void* {
    void* p = ws + off;
    off += (bytes + 255) & ~(size_t)255;
    return p;
  };
  __bf16* xb    = (__bf16*)wsalloc((size_t)MTOT * DIMD * 2);
  __bf16* WqkvT = (__bf16*)wsalloc((size_t)NQKV * DIMD * 2);
  __bf16* WoutT = (__bf16*)wsalloc((size_t)DIMD * DIMD * 2);
  __bf16* Qb    = (__bf16*)wsalloc((size_t)BHTOT * SEQ * DHEAD * 2);
  __bf16* Kb    = (__bf16*)wsalloc((size_t)BHTOT * SEQ * DHEAD * 2);
  __bf16* Vtb   = (__bf16*)wsalloc((size_t)BHTOT * DHEAD * SEQ * 2);
  __bf16* attn  = (__bf16*)wsalloc((size_t)MTOT * DIMD * 2);

  int n8 = MTOT * DIMD / 8;
  k_cvt_bf16<<<(n8 + 255) / 256, 256, 0, stream>>>(x, xb, n8);
  k_transpose_bf16<<<dim3(NQKV / 32, DIMD / 32), 256, 0, stream>>>(Wqkv, WqkvT,
                                                                   DIMD, NQKV);
  k_transpose_bf16<<<dim3(DIMD / 32, DIMD / 32), 256, 0, stream>>>(Wout, WoutT,
                                                                   DIMD, DIMD);
  k_qkv_gemm<<<dim3(NQKV / 128, MTOT / 128), 256, 0, stream>>>(
      xb, WqkvT, bqkv, Qb, Kb, Vtb);
  k_attn<<<BHTOT * 8, 256, 0, stream>>>(Qb, Kb, Vtb, attn);
  k_out_gemm<<<dim3(DIMD / 128, MTOT / 128), 256, 0, stream>>>(attn, WoutT,
                                                               bout, out);
}
